// Int4MatmulCPU_52544629899326
// MI455X (gfx1250) — matrix operand order repaired and checked by reading, not compiled, not run on hardware
//
#include <hip/hip_runtime.h>

#define TOKENS 4096
#define IN_F   4096
#define OUT_F  16384
#define PACKED_DW (IN_F / 8)     // 512 dwords per packed weight row
#define BM 128
#define BN 128
#define BK 32
#define NKT (IN_F / BK)          // 128 k-tiles
#define LDS_ROW_DW 20            // 16 data dwords (32 bf16) + 4 pad dwords

// CDNA5 async global->LDS path (guarded; falls back to reg-staged copy)
#if defined(__has_builtin)
#if __has_builtin(__builtin_amdgcn_global_load_async_to_lds_b128)
#define HAVE_ASYNC_LDS 1
#endif
#endif
#ifndef HAVE_ASYNC_LDS
#define HAVE_ASYNC_LDS 0
#endif

typedef __attribute__((ext_vector_type(16))) __bf16        v16bf;
typedef __attribute__((ext_vector_type(8)))  float         v8f;
typedef __attribute__((ext_vector_type(4)))  unsigned int  uint4v;
typedef __attribute__((ext_vector_type(2)))  unsigned int  uint2v;
typedef __attribute__((ext_vector_type(4)))  float         float4v;

#if HAVE_ASYNC_LDS
// builtin signature (from clang diagnostic): (v4i addrspace(1)*, v4i addrspace(3)*, imm, imm)
typedef __attribute__((ext_vector_type(4))) int i4;
typedef __attribute__((address_space(1))) i4 gas_i4;   // global int4
typedef __attribute__((address_space(3))) i4 lds_i4;   // LDS int4
#endif

union FragAB { uint4v u4[2]; v16bf v; };

__device__ __forceinline__ unsigned short f32_bf16_rne(float f) {
  unsigned u = __builtin_bit_cast(unsigned, f);
  u += 0x7FFFu + ((u >> 16) & 1u);
  return (unsigned short)(u >> 16);
}

// one packed byte -> two bf16 (low nibble = even K, high nibble = odd K), exact
__device__ __forceinline__ unsigned nib_pair_bf16(unsigned byte) {
  float lo = (float)(byte & 15u);
  float hi = (float)((byte >> 4) & 15u);
  return (__builtin_bit_cast(unsigned, lo) >> 16) |
         (__builtin_bit_cast(unsigned, hi) & 0xFFFF0000u);
}

// ---------------- prep: x f32 -> bf16 (ws) + per-token row sums ----------------
__global__ void __launch_bounds__(256) prep_x(const float4v* __restrict__ x4,
                                              uint2v* __restrict__ xbf4,
                                              float* __restrict__ rowsum) {
  const int t   = blockIdx.x;
  const int tid = threadIdx.x;
  const int row4 = IN_F / 4;                 // 1024 float4 per token row
  float s = 0.f;
#pragma unroll
  for (int k = 0; k < 4; ++k) {
    int idx = t * row4 + tid + k * 256;
    float4v v = x4[idx];
    uint2v o;
    o.x = (unsigned)f32_bf16_rne(v.x) | ((unsigned)f32_bf16_rne(v.y) << 16);
    o.y = (unsigned)f32_bf16_rne(v.z) | ((unsigned)f32_bf16_rne(v.w) << 16);
    xbf4[idx] = o;
    s += v.x + v.y + v.z + v.w;
  }
#pragma unroll
  for (int off = 16; off > 0; off >>= 1)
    s += __shfl_down(s, off, 32);
  __shared__ float red[8];
  const int lane = tid & 31, w = tid >> 5;
  if (lane == 0) red[w] = s;
  __syncthreads();
  if (tid == 0) {
    float tot = 0.f;
#pragma unroll
    for (int i = 0; i < 8; ++i) tot += red[i];
    rowsum[t] = tot;
  }
}

// ---------------- main: bf16 WMMA GEMM on raw nibbles, zero/scale in epilogue ----------------
__global__ void __launch_bounds__(256) int4_gemm(
    const unsigned short* __restrict__ xbf,   // [T][K] bf16
    const unsigned int*   __restrict__ wq,    // [O][PACKED_DW] packed nibbles
    const float* __restrict__ scale,          // [O]
    const float* __restrict__ zero,           // [O]
    const float* __restrict__ rowsum,         // [T]
    float* __restrict__ out) {                // [T][O]
  __shared__ unsigned int ldsA[2][BM * LDS_ROW_DW];   // 10 KB each buf
  __shared__ unsigned int ldsB[2][BN * LDS_ROW_DW];

  const int tid   = threadIdx.x;
  const int lane  = tid & 31;
  const int hi    = lane >> 4;          // lane half selects K-subblock
  const int wv    = tid >> 5;           // 0..7
  const int waveM = wv >> 2;            // 0..1  -> 64 token rows each
  const int waveN = wv & 3;             // 0..3  -> 32 out cols each
  const int tileM = blockIdx.y * BM;
  const int tileN = blockIdx.x * BN;

  // staging copy role: row cr (and cr+64), 16B chunk cc
  const int cr = tid >> 2;              // 0..63
  const int cc = tid & 3;               // 0..3

  v8f acc[4][2] = {};

#if !HAVE_ASYNC_LDS
  uint4v stA0, stA1;
#endif
  unsigned stB0, stB1;

  // A tile: issue the global->LDS movement for tile kt into buffer `buf`.
  auto issueA = [&](int kt, int buf) {
    const unsigned short* a0 = xbf + (size_t)(tileM + cr) * IN_F + kt * BK + cc * 8;
#if HAVE_ASYNC_LDS
    __builtin_amdgcn_global_load_async_to_lds_b128(
        (gas_i4*)a0,
        (lds_i4*)&ldsA[buf][cr * LDS_ROW_DW + cc * 4], 0, 0);
    __builtin_amdgcn_global_load_async_to_lds_b128(
        (gas_i4*)(a0 + (size_t)64 * IN_F),
        (lds_i4*)&ldsA[buf][(cr + 64) * LDS_ROW_DW + cc * 4], 0, 0);
#else
    stA0 = *(const uint4v*)a0;
    stA1 = *(const uint4v*)(a0 + (size_t)64 * IN_F);
#endif
  };
  // A tile: make the movement visible in LDS (before the workgroup barrier).
  auto commitA = [&](int buf) {
#if HAVE_ASYNC_LDS
    (void)buf;
#if __has_builtin(__builtin_amdgcn_s_wait_asynccnt)
    __builtin_amdgcn_s_wait_asynccnt(0);
#else
    asm volatile("s_wait_asynccnt 0x0" ::: "memory");
#endif
#else
    *(uint4v*)&ldsA[buf][cr        * LDS_ROW_DW + cc * 4] = stA0;
    *(uint4v*)&ldsA[buf][(cr + 64) * LDS_ROW_DW + cc * 4] = stA1;
#endif
  };

  auto loadB = [&](int kt) {
    const unsigned int* w0 = wq + (size_t)(tileN + cr) * PACKED_DW + kt * 4 + cc;
    stB0 = *w0;
    stB1 = *(w0 + (size_t)64 * PACKED_DW);
  };
  auto storeB = [&](int buf) {
    unsigned int* pb0 = &ldsB[buf][cr        * LDS_ROW_DW + cc * 4];
    unsigned int* pb1 = &ldsB[buf][(cr + 64) * LDS_ROW_DW + cc * 4];
    unsigned b = stB0;
    pb0[0] = nib_pair_bf16(b & 0xFFu);
    pb0[1] = nib_pair_bf16((b >> 8) & 0xFFu);
    pb0[2] = nib_pair_bf16((b >> 16) & 0xFFu);
    pb0[3] = nib_pair_bf16(b >> 24);
    b = stB1;
    pb1[0] = nib_pair_bf16(b & 0xFFu);
    pb1[1] = nib_pair_bf16((b >> 8) & 0xFFu);
    pb1[2] = nib_pair_bf16((b >> 16) & 0xFFu);
    pb1[3] = nib_pair_bf16(b >> 24);
  };

  // prologue: tile 0 into buffer 0
  issueA(0, 0);
  loadB(0);
  storeB(0);
  commitA(0);
  __syncthreads();

  for (int kt = 0; kt < NKT; ++kt) {
    const int cur = kt & 1;
    const int nxt = cur ^ 1;
    if (kt + 1 < NKT) {
      issueA(kt + 1, nxt);   // async writes to LDS[nxt]; prior barrier freed it
      loadB(kt + 1);
    }
    if (kt + 2 < NKT) {
      __builtin_prefetch(xbf + (size_t)(tileM + cr) * IN_F + (kt + 2) * BK + cc * 8, 0, 3);
      __builtin_prefetch(wq + (size_t)(tileN + cr) * PACKED_DW + (kt + 2) * 4 + cc, 0, 3);
    }

    FragAB a[4], b[2];
#pragma unroll
    for (int mt = 0; mt < 4; ++mt) {   // A 16x32 bf16: dwords [hi*4..] K0-7/8-15, [8+hi*4..] K16-23/24-31
      const uint4v* pa =
          (const uint4v*)&ldsA[cur][(waveM * 64 + mt * 16 + (lane & 15)) * LDS_ROW_DW];
      a[mt].u4[0] = pa[hi];
      a[mt].u4[1] = pa[hi + 2];
    }
#pragma unroll
    for (int nt = 0; nt < 2; ++nt) {   // B 32x16 bf16: dwords [hi*4..] K0-7/8-15, [8+hi*4..] K16-23/24-31
      const uint4v* pb =
          (const uint4v*)&ldsB[cur][(waveN * 32 + nt * 16 + (lane & 15)) * LDS_ROW_DW];
      b[nt].u4[0] = pb[hi];
      b[nt].u4[1] = pb[hi + 2];
    }

#pragma unroll
    for (int mt = 0; mt < 4; ++mt)
#pragma unroll
      for (int nt = 0; nt < 2; ++nt)
        acc[mt][nt] = __builtin_amdgcn_wmma_f32_16x16x32_bf16(
            false, a[mt].v, false, b[nt].v, (short)0, acc[mt][nt], false, false);

    if (kt + 1 < NKT) {
      storeB(nxt);
      commitA(nxt);          // s_wait_asynccnt 0 (or ds_store fallback)
    }
    __syncthreads();
  }

  // epilogue: out = scale[o] * (acc - zero[o] * rowsum[t])
  const int nbase = tileN + waveN * 32 + (lane & 15);
  const int mbase = tileM + waveM * 64 + hi * 8;
#pragma unroll
  for (int nt = 0; nt < 2; ++nt) {
    const int o = nbase + nt * 16;
    const float sc = scale[o];
    const float zp = zero[o];
#pragma unroll
    for (int mt = 0; mt < 4; ++mt) {
#pragma unroll
      for (int r = 0; r < 8; ++r) {
        const int m = mbase + mt * 16 + r;
        out[(size_t)m * OUT_F + o] = sc * (acc[mt][nt][r] - zp * rowsum[m]);
      }
    }
  }
}

extern "C" void kernel_launch(void* const* d_in, const int* in_sizes, int n_in,
                              void* d_out, int out_size, void* d_ws, size_t ws_size,
                              hipStream_t stream) {
  const float*        x     = (const float*)d_in[0];
  const unsigned int* wq    = (const unsigned int*)d_in[1];
  const float*        scale = (const float*)d_in[2];
  const float*        zero  = (const float*)d_in[3];

  unsigned short* xbf    = (unsigned short*)d_ws;                       // 32 MB
  float*          rowsum = (float*)((char*)d_ws + (size_t)TOKENS * IN_F * 2);

  prep_x<<<TOKENS, 256, 0, stream>>>((const float4v*)x, (uint2v*)xbf, rowsum);

  dim3 grid(OUT_F / BN, TOKENS / BM);
  int4_gemm<<<grid, 256, 0, stream>>>(xbf, wq, scale, zero, rowsum, (float*)d_out);
}
